// BiMambaLayer_53515292508674
// MI455X (gfx1250) — compile-verified, matrix operand order checked by reading, not run on hardware
//
#include <hip/hip_runtime.h>
#include <cstdint>
#include <cstddef>

// ---------------------------------------------------------------------------
// BiMamba layer for MI455X (gfx1250, wave32, WMMA).
// All GEMMs run through v_wmma_f32_16x16x32_bf16 (bf16 in / f32 accum).
// Roofline: ~68 GFLOP of GEMM vs ~300MB traffic @23.3TB/s -> compute-side,
// so everything dense is routed to the matrix pipe; the sequential selective
// scan stays in f32 VALU with LDS-broadcast B/C per timestep.
// Tile staging uses GLOBAL_LOAD_ASYNC_TO_LDS_B128 (ASYNCcnt) when the
// toolchain exposes the builtin; otherwise falls back to reg staging.
// ---------------------------------------------------------------------------

typedef unsigned short u16;
typedef __attribute__((ext_vector_type(16))) __bf16 v16bf;
typedef __attribute__((ext_vector_type(8)))  float  v8f;
typedef __attribute__((ext_vector_type(4)))  int    v4i;

union FragBF { v16bf v; uint4 q[2]; };

#define BATCH  4
#define SEQ    1024
#define DM     512
#define ED     1024
#define NSTATE 16
#define DTRANK 32
#define MTOK   4096   // BATCH*SEQ (rows per direction)
#define DFF    1536

#if defined(__has_builtin)
#if __has_builtin(__builtin_amdgcn_global_load_async_to_lds_b128) && \
    __has_builtin(__builtin_amdgcn_s_wait_asynccnt)
#define USE_ASYNC_LDS 1
#endif
#endif
#ifndef USE_ASYNC_LDS
#define USE_ASYNC_LDS 0
#endif

__device__ __forceinline__ u16 f2bf(float f) {
  unsigned u = __float_as_uint(f);
  unsigned r = u + 0x7FFFu + ((u >> 16) & 1u);   // round-to-nearest-even
  return (u16)(r >> 16);
}
__device__ __forceinline__ float sigmoidf_(float x) { return 1.f / (1.f + __expf(-x)); }
__device__ __forceinline__ float siluf_(float x)    { return x * sigmoidf_(x); }

#if USE_ASYNC_LDS
// 16-byte async copy global -> LDS (tracked by ASYNCcnt, no VGPR round-trip).
// Builtin signature (probed from hipcc diagnostic): (v4i AS1*, v4i AS3*, imm, imm)
__device__ __forceinline__ void async_cp16(const u16* g, u16* l) {
  __builtin_amdgcn_global_load_async_to_lds_b128(
      (__attribute__((address_space(1))) v4i*)(uintptr_t)g,
      (__attribute__((address_space(3))) v4i*)(uintptr_t)l,
      0, 0);
}
#endif

// ---------------------------------------------------------------------------
// Generic WMMA GEMM: out(M,N) = A(M,K)[bf16] * W(N,K)^T[bf16] (+bias)
// Block tile 128x64, BK=32 (one WMMA K-step per tile), 8 waves -> 32x32/wave.
// EPI 0: f32 store            (ldout, +bias optional)
// EPI 1: bf16 store           (+col_off, optional seq-flip of row: m^1023)
// EPI 2: f32 residual store   out = resid + ls[n]*(acc+bias)
// ---------------------------------------------------------------------------
template <int EPI>
__global__ __launch_bounds__(256) void gemm_bf16_wmma(
    const u16* __restrict__ A, const u16* __restrict__ B,
    float* __restrict__ outf, u16* __restrict__ outb,
    const float* __restrict__ bias, const float* __restrict__ resid,
    const float* __restrict__ ls,
    int M, int N, int K, int ldout, int col_off, int flip)
{
  __shared__ u16 lsA[2][128 * 32];
  __shared__ u16 lsB[2][64 * 32];

  const int tid  = threadIdx.x;
  const int lane = tid & 31;
  const int wave = tid >> 5;
  const int half = lane >> 4;       // K-half selector per ISA 16-bit layout
  const int lr   = lane & 15;       // row (A) / col (B) within fragment
  const int wm   = (wave >> 1) * 32;
  const int wn   = (wave & 1) * 32;
  const int m0   = blockIdx.y * 128;
  const int n0   = blockIdx.x * 64;
  const int nk   = K >> 5;

  const int lrow0 = tid >> 2;            // 0..63
  const int lrow1 = (tid + 256) >> 2;    // 64..127
  const int lkv   = (tid & 3) * 8;       // bf16 element offset within BK row

  // Stage one 128x32 A tile + 64x32 B tile into LDS buffer `buf`.
  auto stage_tile = [&](int kt, int buf) {
    const size_t ko = (size_t)kt * 32;
#if USE_ASYNC_LDS
    async_cp16(A + (size_t)(m0 + lrow0) * K + ko + lkv, &lsA[buf][lrow0 * 32 + lkv]);
    async_cp16(A + (size_t)(m0 + lrow1) * K + ko + lkv, &lsA[buf][lrow1 * 32 + lkv]);
    async_cp16(B + (size_t)(n0 + lrow0) * K + ko + lkv, &lsB[buf][lrow0 * 32 + lkv]);
#else
    uint4 a0 = *(const uint4*)(A + (size_t)(m0 + lrow0) * K + ko + lkv);
    uint4 a1 = *(const uint4*)(A + (size_t)(m0 + lrow1) * K + ko + lkv);
    uint4 b0 = *(const uint4*)(B + (size_t)(n0 + lrow0) * K + ko + lkv);
    *(uint4*)&lsA[buf][lrow0 * 32 + lkv] = a0;
    *(uint4*)&lsA[buf][lrow1 * 32 + lkv] = a1;
    *(uint4*)&lsB[buf][lrow0 * 32 + lkv] = b0;
#endif
  };

  // ---- stage tile 0 ----
  stage_tile(0, 0);
#if USE_ASYNC_LDS
  __builtin_amdgcn_s_wait_asynccnt(0);
#endif
  __syncthreads();

  v8f acc[2][2];
#pragma unroll
  for (int i = 0; i < 2; ++i)
#pragma unroll
    for (int j = 0; j < 2; ++j) { v8f z = {}; acc[i][j] = z; }

  for (int kt = 0; kt < nk; ++kt) {
    const int cur = kt & 1;
    if (kt + 1 < nk)            // stage next K-tile into other buffer
      stage_tile(kt + 1, cur ^ 1);
    if (kt + 2 < nk) {          // L2 prefetch (global_prefetch_b8 on gfx1250)
      __builtin_prefetch(A + (size_t)(m0 + lrow0) * K + (size_t)(kt + 2) * 32 + lkv, 0, 0);
      __builtin_prefetch(B + (size_t)(n0 + lrow0) * K + (size_t)(kt + 2) * 32 + lkv, 0, 0);
    }

    // ---- fragments per ISA 16-bit A/B VGPR layout ----
    FragBF af[2], bfm[2];
#pragma unroll
    for (int f = 0; f < 2; ++f) {
      const int ra = (wm + f * 16 + lr) * 32 + half * 8;
      af[f].q[0] = *(const uint4*)&lsA[cur][ra];
      af[f].q[1] = *(const uint4*)&lsA[cur][ra + 16];
      const int rb = (wn + f * 16 + lr) * 32 + half * 8;
      bfm[f].q[0] = *(const uint4*)&lsB[cur][rb];
      bfm[f].q[1] = *(const uint4*)&lsB[cur][rb + 16];
    }
#pragma unroll
    for (int i = 0; i < 2; ++i)
#pragma unroll
      for (int j = 0; j < 2; ++j)
        acc[i][j] = __builtin_amdgcn_wmma_f32_16x16x32_bf16(
            false, af[i].v, false, bfm[j].v, (short)0, acc[i][j], false, false);

    if (kt + 1 < nk) {
#if USE_ASYNC_LDS
      __builtin_amdgcn_s_wait_asynccnt(0);
#endif
      __syncthreads();
    }
  }

  // ---- epilogue (C/D layout: lane = N col, VGPR r = M row (+8 for hi half)) ----
#pragma unroll
  for (int i = 0; i < 2; ++i)
#pragma unroll
    for (int j = 0; j < 2; ++j) {
      const int mbase = m0 + wm + i * 16 + half * 8;
      const int nc    = n0 + wn + j * 16 + lr;
      const float bv  = bias ? bias[nc] : 0.f;
#pragma unroll
      for (int r = 0; r < 8; ++r) {
        const int m = mbase + r;
        float val = acc[i][j][r] + bv;
        if (EPI == 0) {
          outf[(size_t)m * ldout + nc] = val;
        } else if (EPI == 1) {
          const int mo = flip ? (m ^ (SEQ - 1)) : m;   // flip l within batch
          outb[(size_t)mo * ldout + col_off + nc] = f2bf(val);
        } else {
          outf[(size_t)m * ldout + nc] =
              resid[(size_t)m * ldout + nc] + ls[nc] * val;
        }
      }
    }
}

// ---------------------------------------------------------------------------
// Elementwise / small kernels
// ---------------------------------------------------------------------------
__global__ __launch_bounds__(256) void cvt_bf16_kernel(
    const float* __restrict__ src, u16* __restrict__ dst, int n)
{
  int i = blockIdx.x * 256 + threadIdx.x;
  if (i < n) dst[i] = f2bf(src[i]);
}

// s = silu(phys @ w1^T + b1)  (K=2, pure elementwise)
__global__ __launch_bounds__(256) void cond_mlp_kernel(
    const float* __restrict__ phys, const float* __restrict__ w1,
    const float* __restrict__ b1, u16* __restrict__ s)
{
  int i = blockIdx.x * 256 + threadIdx.x;    // MTOK*1024
  int t = i >> 10, j = i & 1023;
  float c0 = phys[(size_t)t * 2], c1 = phys[(size_t)t * 2 + 1];
  float v = c0 * w1[j * 2] + c1 * w1[j * 2 + 1] + b1[j];
  s[i] = f2bf(siluf_(v));
}

// AdaNorm apply: h = x/rms*(scale+tanh(g)*0.5)+tanh(b)*0.5 -> bf16 (+flipped)
__global__ __launch_bounds__(256) void adanorm_kernel(
    const float* __restrict__ x, const float* __restrict__ gb,
    const float* __restrict__ scale, u16* __restrict__ hout,
    u16* __restrict__ hflip, int do_flip)
{
  const int t = blockIdx.x, tid = threadIdx.x;
  __shared__ float red[256];
  const float x0 = x[(size_t)t * DM + tid];
  const float x1 = x[(size_t)t * DM + 256 + tid];
  red[tid] = x0 * x0 + x1 * x1;
  __syncthreads();
  for (int s = 128; s > 0; s >>= 1) {
    if (tid < s) red[tid] += red[tid + s];
    __syncthreads();
  }
  const float inv = rsqrtf(red[0] * (1.f / DM) + 1e-6f);
  const float xv[2] = {x0, x1};
#pragma unroll
  for (int p = 0; p < 2; ++p) {
    const int c = tid + p * 256;
    const float g  = tanhf(gb[(size_t)t * 1024 + c]) * 0.5f;
    const float be = tanhf(gb[(size_t)t * 1024 + 512 + c]) * 0.5f;
    const u16 hv = f2bf(xv[p] * inv * (scale[c] + g) + be);
    hout[(size_t)t * DM + c] = hv;
    if (do_flip) hflip[(size_t)(t ^ (SEQ - 1)) * DM + c] = hv;
  }
}

// depthwise causal conv (d_conv=4) + SiLU over xs half of xz
__global__ __launch_bounds__(256) void conv_silu_kernel(
    const float* __restrict__ xz, const float* __restrict__ cw,
    const float* __restrict__ cb, float* __restrict__ uf, u16* __restrict__ ub)
{
  int i = blockIdx.x * 256 + threadIdx.x;    // 2*MTOK*ED
  int e = i & (ED - 1);
  int m = (i >> 10) & (MTOK - 1);
  int dir = i >> 22;
  int l = m & (SEQ - 1), bb = m >> 10;
  const float* xzd = xz + (size_t)dir * MTOK * (2 * ED);
  float acc = cb[dir * ED + e];
#pragma unroll
  for (int k = 0; k < 4; ++k) {
    int l2 = l + k - 3;
    if (l2 >= 0)
      acc += cw[((size_t)dir * ED + e) * 4 + k] *
             xzd[(size_t)(bb * SEQ + l2) * (2 * ED) + e];
  }
  float u = siluf_(acc);
  uf[i] = u;
  ub[i] = f2bf(u);
}

// dtr (first 32 cols of dbc) -> bf16, contiguous (M,32) for the dt GEMM
__global__ __launch_bounds__(256) void extract_dtr_kernel(
    const float* __restrict__ dbc, u16* __restrict__ dtr)
{
  int i = blockIdx.x * 256 + threadIdx.x;    // 2*MTOK*32
  int r = i & 31, m = i >> 5;
  dtr[i] = f2bf(dbc[(size_t)m * 64 + r]);
}

// Selective scan: block = (dir, b, 256-channel chunk); thread = one channel e.
// B/C (32 floats/timestep, shared by all channels) double-buffered in LDS.
__global__ __launch_bounds__(256) void scan_kernel(
    const float* __restrict__ dbc,   // (2,MTOK,64): [.,32:48]=B, [.,48:64]=C
    const float* __restrict__ dtp,   // (2,MTOK,ED) pre-softplus dt
    const float* __restrict__ uf,    // (2,MTOK,ED)
    const float* __restrict__ xz,    // (2,MTOK,2*ED), z in cols [ED,2ED)
    const float* __restrict__ A_log, // (2,ED,16)
    const float* __restrict__ Dsk,   // (2,ED)
    u16* __restrict__ yg)            // (2,MTOK,ED) bf16: (y+u*D)*silu(z)
{
  const int tid = threadIdx.x;
  const int bid = blockIdx.x;        // 0..31
  const int dir = bid >> 4;
  const int b   = (bid >> 2) & 3;
  const int e   = (bid & 3) * 256 + tid;

  const float* dbc_d = dbc + (size_t)dir * MTOK * 64;
  const float* dtp_d = dtp + (size_t)dir * MTOK * ED;
  const float* uf_d  = uf  + (size_t)dir * MTOK * ED;
  const float* xz_d  = xz  + (size_t)dir * MTOK * (2 * ED);
  u16* yg_d          = yg  + (size_t)dir * MTOK * ED;

  float a[NSTATE];
#pragma unroll
  for (int n = 0; n < NSTATE; ++n)
    a[n] = -__expf(A_log[((size_t)dir * ED + e) * NSTATE + n]);
  const float Dv = Dsk[dir * ED + e];

  float h[NSTATE];
#pragma unroll
  for (int n = 0; n < NSTATE; ++n) h[n] = 0.f;

  __shared__ float sBC[2][32];
  for (int l = 0; l < SEQ; ++l) {
    const int m  = b * SEQ + l;
    const int pb = l & 1;
    if (tid < 32) sBC[pb][tid] = dbc_d[(size_t)m * 64 + 32 + tid];
    __syncthreads();

    const float dpre = dtp_d[(size_t)m * ED + e];
    const float dt = (dpre > 20.f) ? dpre : log1pf(__expf(dpre));
    const float uv = uf_d[(size_t)m * ED + e];
    const float du = dt * uv;
    float y = 0.f;
#pragma unroll
    for (int n = 0; n < NSTATE; ++n) {
      const float da = __expf(dt * a[n]);
      h[n] = da * h[n] + du * sBC[pb][n];
      y += h[n] * sBC[pb][16 + n];
    }
    const float zv = xz_d[(size_t)m * (2 * ED) + ED + e];
    yg_d[(size_t)m * ED + e] = f2bf((y + uv * Dv) * siluf_(zv));
  }
}

// SwiGLU gate: ffin = silu(gv[:,:1536]) * gv[:,1536:]
__global__ __launch_bounds__(256) void gate_kernel(
    const float* __restrict__ gv, u16* __restrict__ ffin)
{
  int i = blockIdx.x * 256 + threadIdx.x;   // MTOK*DFF
  int m = i / DFF, j = i - m * DFF;
  float g = gv[(size_t)m * (2 * DFF) + j];
  float v = gv[(size_t)m * (2 * DFF) + DFF + j];
  ffin[i] = f2bf(siluf_(g) * v);
}

// ---------------------------------------------------------------------------
// Host driver
// ---------------------------------------------------------------------------
extern "C" void kernel_launch(void* const* d_in, const int* in_sizes, int n_in,
                              void* d_out, int out_size, void* d_ws, size_t ws_size,
                              hipStream_t stream)
{
  (void)in_sizes; (void)n_in; (void)out_size; (void)ws_size;

  const float* x        = (const float*)d_in[0];
  const float* phys     = (const float*)d_in[1];
  const float* n1_scale = (const float*)d_in[2];
  const float* n1_w1    = (const float*)d_in[3];
  const float* n1_b1    = (const float*)d_in[4];
  const float* n1_w2    = (const float*)d_in[5];
  const float* n1_b2    = (const float*)d_in[6];
  const float* n2_scale = (const float*)d_in[7];
  const float* n2_w1    = (const float*)d_in[8];
  const float* n2_b1    = (const float*)d_in[9];
  const float* n2_w2    = (const float*)d_in[10];
  const float* n2_b2    = (const float*)d_in[11];
  const float* m_in_w   = (const float*)d_in[12];
  const float* m_conv_w = (const float*)d_in[13];
  const float* m_conv_b = (const float*)d_in[14];
  const float* m_xproj  = (const float*)d_in[15];
  const float* m_dt_w   = (const float*)d_in[16];
  const float* m_dt_b   = (const float*)d_in[17];
  const float* m_A_log  = (const float*)d_in[18];
  const float* m_D      = (const float*)d_in[19];
  const float* m_out_w  = (const float*)d_in[20];
  const float* blk_w    = (const float*)d_in[21];
  const float* blk_b    = (const float*)d_in[22];
  const float* fc1_w    = (const float*)d_in[23];
  const float* fc1_b    = (const float*)d_in[24];
  const float* fc2_w    = (const float*)d_in[25];
  const float* fc2_b    = (const float*)d_in[26];
  const float* ls1      = (const float*)d_in[27];
  const float* ls2      = (const float*)d_in[28];
  float* out = (float*)d_out;

  // ---- scratch carve-out (~290MB) ----
  char* wsb = (char*)d_ws;
  size_t off = 0;
  auto alloc = [&](size_t bytes) -> char* {
    char* p = wsb + off;
    off += (bytes + 255) & ~(size_t)255;
    return p;
  };
  u16*   wb_n1w2 = (u16*)alloc((size_t)1024 * 1024 * 2);
  u16*   wb_n2w2 = (u16*)alloc((size_t)1024 * 1024 * 2);
  u16*   wb_inw  = (u16*)alloc((size_t)2 * 2048 * 512 * 2);
  u16*   wb_xprj = (u16*)alloc((size_t)2 * 64 * 1024 * 2);
  u16*   wb_dtw  = (u16*)alloc((size_t)2 * 1024 * 32 * 2);
  u16*   wb_outw = (u16*)alloc((size_t)2 * 512 * 1024 * 2);
  u16*   wb_blkw = (u16*)alloc((size_t)512 * 1024 * 2);
  u16*   wb_fc1w = (u16*)alloc((size_t)3072 * 512 * 2);
  u16*   wb_fc2w = (u16*)alloc((size_t)512 * 1536 * 2);
  u16*   s_bf    = (u16*)alloc((size_t)MTOK * 1024 * 2);
  float* gb      = (float*)alloc((size_t)MTOK * 1024 * 4);
  u16*   h_bf    = (u16*)alloc((size_t)2 * MTOK * DM * 2);
  float* xz      = (float*)alloc((size_t)2 * MTOK * 2048 * 4);
  float* u_f     = (float*)alloc((size_t)2 * MTOK * ED * 4);
  u16*   u_bf    = (u16*)alloc((size_t)2 * MTOK * ED * 2);
  float* dbc     = (float*)alloc((size_t)2 * MTOK * 64 * 4);
  u16*   dtr_bf  = (u16*)alloc((size_t)2 * MTOK * 32 * 2);
  float* dtp     = (float*)alloc((size_t)2 * MTOK * ED * 4);
  u16*   yg_bf   = (u16*)alloc((size_t)2 * MTOK * ED * 2);
  u16*   ycat    = (u16*)alloc((size_t)MTOK * 1024 * 2);
  float* x1      = (float*)alloc((size_t)MTOK * DM * 4);
  float* gv      = (float*)alloc((size_t)MTOK * 3072 * 4);
  u16*   ffin    = (u16*)alloc((size_t)MTOK * DFF * 2);

  auto cvt = [&](const float* src, u16* dst, size_t n) {
    cvt_bf16_kernel<<<dim3((unsigned)((n + 255) / 256)), dim3(256), 0, stream>>>(
        src, dst, (int)n);
  };
  auto gemm = [&](int epi, const u16* A, const u16* B, float* of, u16* ob,
                  const float* bias, const float* resid, const float* lsv,
                  int M, int N, int K, int ldo, int coff, int flip) {
    dim3 g((unsigned)(N / 64), (unsigned)(M / 128)), blk(256);
    if (epi == 0)
      gemm_bf16_wmma<0><<<g, blk, 0, stream>>>(A, B, of, ob, bias, resid, lsv,
                                               M, N, K, ldo, coff, flip);
    else if (epi == 1)
      gemm_bf16_wmma<1><<<g, blk, 0, stream>>>(A, B, of, ob, bias, resid, lsv,
                                               M, N, K, ldo, coff, flip);
    else
      gemm_bf16_wmma<2><<<g, blk, 0, stream>>>(A, B, of, ob, bias, resid, lsv,
                                               M, N, K, ldo, coff, flip);
  };

  // ---- weights -> bf16 ----
  cvt(n1_w2,   wb_n1w2, (size_t)1024 * 1024);
  cvt(n2_w2,   wb_n2w2, (size_t)1024 * 1024);
  cvt(m_in_w,  wb_inw,  (size_t)2 * 2048 * 512);
  cvt(m_xproj, wb_xprj, (size_t)2 * 64 * 1024);
  cvt(m_dt_w,  wb_dtw,  (size_t)2 * 1024 * 32);
  cvt(m_out_w, wb_outw, (size_t)2 * 512 * 1024);
  cvt(blk_w,   wb_blkw, (size_t)512 * 1024);
  cvt(fc1_w,   wb_fc1w, (size_t)3072 * 512);
  cvt(fc2_w,   wb_fc2w, (size_t)512 * 1536);

  // ---- AdaNorm 1 ----
  cond_mlp_kernel<<<MTOK * 1024 / 256, 256, 0, stream>>>(phys, n1_w1, n1_b1, s_bf);
  gemm(0, s_bf, wb_n1w2, gb, nullptr, n1_b2, nullptr, nullptr,
       MTOK, 1024, 1024, 1024, 0, 0);
  adanorm_kernel<<<MTOK, 256, 0, stream>>>(x, gb, n1_scale, h_bf,
                                           h_bf + (size_t)MTOK * DM, 1);

  // ---- in_proj per direction ----
  for (int d = 0; d < 2; ++d)
    gemm(0, h_bf + (size_t)d * MTOK * DM, wb_inw + (size_t)d * 2048 * 512,
         xz + (size_t)d * MTOK * 2048, nullptr, nullptr, nullptr, nullptr,
         MTOK, 2048, 512, 2048, 0, 0);

  // ---- depthwise conv + SiLU ----
  conv_silu_kernel<<<(2 * MTOK * ED) / 256, 256, 0, stream>>>(
      xz, m_conv_w, m_conv_b, u_f, u_bf);

  // ---- x_proj (N=64) ----
  for (int d = 0; d < 2; ++d)
    gemm(0, u_bf + (size_t)d * MTOK * ED, wb_xprj + (size_t)d * 64 * 1024,
         dbc + (size_t)d * MTOK * 64, nullptr, nullptr, nullptr, nullptr,
         MTOK, 64, 1024, 64, 0, 0);

  // ---- dt projection (K=32, single WMMA step) ----
  extract_dtr_kernel<<<(2 * MTOK * 32) / 256, 256, 0, stream>>>(dbc, dtr_bf);
  for (int d = 0; d < 2; ++d)
    gemm(0, dtr_bf + (size_t)d * MTOK * 32, wb_dtw + (size_t)d * 1024 * 32,
         dtp + (size_t)d * MTOK * ED, nullptr, m_dt_b + d * 1024, nullptr,
         nullptr, MTOK, 1024, 32, 1024, 0, 0);

  // ---- selective scan (writes (y+u*D)*silu(z) as bf16) ----
  scan_kernel<<<32, 256, 0, stream>>>(dbc, dtp, u_f, xz, m_A_log, m_D, yg_bf);

  // ---- out_proj -> concatenated ycat (dir1 flipped back, cols 512..1023) ----
  for (int d = 0; d < 2; ++d)
    gemm(1, yg_bf + (size_t)d * MTOK * ED, wb_outw + (size_t)d * 512 * 1024,
         nullptr, ycat, nullptr, nullptr, nullptr,
         MTOK, 512, 1024, 1024, d * 512, d);

  // ---- block proj + residual: x1 = x + ls1*(ycat@blk_w^T + blk_b) ----
  gemm(2, ycat, wb_blkw, x1, nullptr, blk_b, x, ls1,
       MTOK, 512, 1024, 512, 0, 0);

  // ---- AdaNorm 2 ----
  cond_mlp_kernel<<<MTOK * 1024 / 256, 256, 0, stream>>>(phys, n2_w1, n2_b1, s_bf);
  gemm(0, s_bf, wb_n2w2, gb, nullptr, n2_b2, nullptr, nullptr,
       MTOK, 1024, 1024, 1024, 0, 0);
  adanorm_kernel<<<MTOK, 256, 0, stream>>>(x1, gb, n2_scale, h_bf, nullptr, 0);

  // ---- FFN ----
  gemm(0, h_bf, wb_fc1w, gv, nullptr, fc1_b, nullptr, nullptr,
       MTOK, 3072, 512, 3072, 0, 0);
  gate_kernel<<<(MTOK * DFF) / 256, 256, 0, stream>>>(gv, ffin);
  gemm(2, ffin, wb_fc2w, out, nullptr, fc2_b, x1, ls2,
       MTOK, 512, 1536, 512, 0, 0);
}
